// MultiHeadSplitAttentionSpatioToken_19035295056360
// MI455X (gfx1250) — compile-verified, matrix operand order checked by reading, not run on hardware
//
#include <hip/hip_runtime.h>
#include <hip/hip_fp16.h>

// ---------------------------------------------------------------------------
// TimeSformer divided space-time attention for MI455X (gfx1250, wave32, WMMA)
// f16 WMMA (fp32 acc) for all GEMMs; fused LDS-resident spatial attention.
// Fragments load as contiguous ds_load_b128 pairs; GEMM global->LDS staging
// is register-pipelined so loads overlap WMMA compute.
// ---------------------------------------------------------------------------

typedef __attribute__((ext_vector_type(16))) _Float16 v16h;
typedef __attribute__((ext_vector_type(8)))  float    v8f;

#define T_DIM  8
#define B_DIM  16
#define NHEAD  12
#define DH     64
#define C_DIM  768
#define WH1    197
#define SEQ    (T_DIM * WH1)            // 1576
#define MROWS  (SEQ * B_DIM)            // 25216
#define NPAD   224                      // 197 -> 14 WMMA col tiles
#define XPAD   208                      // 197 -> 13 row tiles (16B-aligned rows)
#define G_SPA  (NHEAD * B_DIM * T_DIM)  // 1536

// -------------------- fragment loaders -------------------------------------
// 16-bit 16x16x32 fragment: lane(half=lane>>4, mrow=lane&15) holds two
// contiguous 8-half runs: K = half*8 + [0..7] and K = 16 + half*8 + [0..7].
// Row-major source (ld halfs per row) => exactly two b128 LDS loads.

__device__ __forceinline__ v16h load_frag16(const _Float16* base, int ld,
                                            int row, int k0, int half) {
    const _Float16* p = base + row * ld + k0 + half * 8;
    v16h a;
    int4* ap = reinterpret_cast<int4*>(&a);
    ap[0] = *reinterpret_cast<const int4*>(p);
    ap[1] = *reinterpret_cast<const int4*>(p + 16);
    return a;
}

// fp32 source (P matrix): four b128 loads + cvt to f16
__device__ __forceinline__ v16h load_frag16_f32(const float* base, int ld,
                                                int row, int k0, int half) {
    const float* p = base + row * ld + k0 + half * 8;
    float4 l0 = *reinterpret_cast<const float4*>(p);
    float4 l1 = *reinterpret_cast<const float4*>(p + 4);
    float4 h0 = *reinterpret_cast<const float4*>(p + 16);
    float4 h1 = *reinterpret_cast<const float4*>(p + 20);
    v16h a;
    a[0]=(_Float16)l0.x;  a[1]=(_Float16)l0.y;  a[2]=(_Float16)l0.z;  a[3]=(_Float16)l0.w;
    a[4]=(_Float16)l1.x;  a[5]=(_Float16)l1.y;  a[6]=(_Float16)l1.z;  a[7]=(_Float16)l1.w;
    a[8]=(_Float16)h0.x;  a[9]=(_Float16)h0.y;  a[10]=(_Float16)h0.z; a[11]=(_Float16)h0.w;
    a[12]=(_Float16)h1.x; a[13]=(_Float16)h1.y; a[14]=(_Float16)h1.z; a[15]=(_Float16)h1.w;
    return a;
}

__device__ __forceinline__ v8f wmma_f16(v16h a, v16h b, v8f c) {
    return __builtin_amdgcn_wmma_f32_16x16x32_f16(false, a, false, b,
                                                  (short)0, c, false, false);
}

// -------------------- conversion kernel ------------------------------------

__global__ void cvt_f32_to_f16(const float* __restrict__ src,
                               _Float16* __restrict__ dst, int n) {
    int i = blockIdx.x * blockDim.x + threadIdx.x;
    if (i < n) dst[i] = (_Float16)src[i];
}

// -------------------- WMMA GEMM: C = A(MxK) * Bt(NxK)^T + bias -------------
// Bt is the weight matrix as stored: [N][K] row-major = transposed-B layout
// the fragment loader wants. 256 threads = 8 waves; tile 128(M) x 64(N);
// K stepped 32 through LDS with register-pipelined staging (next K-tile's
// global loads issued before the WMMA work on the current tile).
// mode 0: fp32 out (out proj).  mode 1: f16 split Q/K/V (QKV proj).

__global__ void gemm_wmma_f16(const _Float16* __restrict__ A,
                              const _Float16* __restrict__ Bt,
                              const float* __restrict__ bias,
                              float* __restrict__ outF,
                              _Float16* __restrict__ outQ,
                              _Float16* __restrict__ outK,
                              _Float16* __restrict__ outV,
                              int M, int N, int K, int mode) {
    __shared__ __align__(16) _Float16 As[128 * 32];   // 8 KB
    __shared__ __align__(16) _Float16 Bs[64 * 32];    // 4 KB, [n][k]
    const int tid  = threadIdx.x;
    const int wave = tid >> 5;
    const int lane = tid & 31;
    const int half = lane >> 4;
    const int mrow = lane & 15;
    const int mBase = blockIdx.y * 128;
    const int nBase = blockIdx.x * 64;

    // each thread stages 2 A-chunks + 1 B-chunk (int4) per K step
    const int ar0 = tid >> 2;                 // A rows 0..63
    const int ar1 = (tid >> 2) + 64;          // A rows 64..127
    const int ac  = tid & 3;                  // 8-half column chunk
    const _Float16* aSrc0 = A  + (size_t)(mBase + ar0) * K + ac * 8;
    const _Float16* aSrc1 = A  + (size_t)(mBase + ar1) * K + ac * 8;
    const _Float16* bSrc  = Bt + (size_t)(nBase + ar0) * K + ac * 8;

    v8f acc0 = {}, acc1 = {}, acc2 = {}, acc3 = {};

    // preload first K tile into registers
    int4 ra0 = *(const int4*)(aSrc0);
    int4 ra1 = *(const int4*)(aSrc1);
    int4 rb0 = *(const int4*)(bSrc);

    for (int k0 = 0; k0 < K; k0 += 32) {
        ((int4*)As)[tid]       = ra0;
        ((int4*)As)[tid + 256] = ra1;
        ((int4*)Bs)[tid]       = rb0;
        __syncthreads();

        if (k0 + 32 < K) {   // issue next tile's loads; overlap with WMMA
            ra0 = *(const int4*)(aSrc0 + k0 + 32);
            ra1 = *(const int4*)(aSrc1 + k0 + 32);
            rb0 = *(const int4*)(bSrc  + k0 + 32);
            if (k0 + 64 < K) __builtin_prefetch(aSrc0 + k0 + 64, 0, 1);
        }

        v16h a = load_frag16(As, 32, wave * 16 + mrow, 0, half);
        v16h b;
        b = load_frag16(Bs, 32,  0 + mrow, 0, half); acc0 = wmma_f16(a, b, acc0);
        b = load_frag16(Bs, 32, 16 + mrow, 0, half); acc1 = wmma_f16(a, b, acc1);
        b = load_frag16(Bs, 32, 32 + mrow, 0, half); acc2 = wmma_f16(a, b, acc2);
        b = load_frag16(Bs, 32, 48 + mrow, 0, half); acc3 = wmma_f16(a, b, acc3);
        __syncthreads();
    }

#pragma unroll
    for (int ct = 0; ct < 4; ++ct) {
        v8f acc = (ct == 0) ? acc0 : (ct == 1) ? acc1 : (ct == 2) ? acc2 : acc3;
        int gn = nBase + ct * 16 + mrow;
#pragma unroll
        for (int i = 0; i < 8; ++i) {
            int gm = mBase + wave * 16 + i + 8 * half;
            if (gm < M && gn < N) {
                float v = acc[i] + bias[gn];
                if (mode == 0) {
                    outF[(size_t)gm * N + gn] = v;
                } else {
                    int part = gn / C_DIM, c = gn % C_DIM;
                    _Float16* p = (part == 0) ? outQ : (part == 1) ? outK : outV;
                    p[(size_t)gm * C_DIM + c] = (_Float16)v;
                }
            }
        }
    }
}

// -------------------- temporal attention (T=8, per (h,b,w) group) ----------
// VALU kernel (8x8 scores too small for 16x16 WMMA tiles). Emits out_t in
// [g][d][x(pad 208)] layout = transposed V for the spatial WMMA pass.

__global__ void temporal_attn(const _Float16* __restrict__ Qh,
                              const _Float16* __restrict__ Kh,
                              const _Float16* __restrict__ Vh,
                              _Float16* __restrict__ outT) {
    int g = blockIdx.x;
    int w = g % WH1;
    int b = (g / WH1) % B_DIM;
    int h = g / (WH1 * B_DIM);

    __shared__ float qs[T_DIM][DH], ks[T_DIM][DH], vs[T_DIM][DH];
    __shared__ float sc[T_DIM][T_DIM];
    int d = threadIdx.x;  // 0..63

    for (int t = 0; t < T_DIM; ++t) {
        size_t base = ((size_t)((t * WH1 + w) * B_DIM + b)) * C_DIM + h * DH + d;
        qs[t][d] = (float)Qh[base];
        ks[t][d] = (float)Kh[base];
        vs[t][d] = (float)Vh[base];
    }
    __syncthreads();

    {   // one (t,s) score per thread
        int t = d >> 3, s = d & 7;
        float acc = 0.f;
        for (int dd = 0; dd < DH; ++dd) acc += qs[t][dd] * ks[s][dd];
        sc[t][s] = acc * 0.125f;  // 1/sqrt(64)
    }
    __syncthreads();

    if (d < T_DIM) {  // row softmax
        float m = -1e30f;
        for (int s = 0; s < T_DIM; ++s) m = fmaxf(m, sc[d][s]);
        float sum = 0.f, e[T_DIM];
        for (int s = 0; s < T_DIM; ++s) { e[s] = __expf(sc[d][s] - m); sum += e[s]; }
        float inv = 1.f / sum;
        for (int s = 0; s < T_DIM; ++s) sc[d][s] = e[s] * inv;
    }
    __syncthreads();

    // out_t: [g' = (h*16+b)*8+t][d][x=w], x padded to 208
    for (int t = 0; t < T_DIM; ++t) {
        float acc = 0.f;
        for (int s = 0; s < T_DIM; ++s) acc += sc[t][s] * vs[s][d];
        size_t o = ((size_t)((h * B_DIM + b) * T_DIM + t)) * DH * XPAD
                   + (size_t)d * XPAD + w;
        outT[o] = (_Float16)acc;
    }
}

// -------------------- fused spatial attention ------------------------------
// One workgroup (8 waves) per g=(h,b,t). LDS: Q [208][64], K [224][64]
// (= transposed-B for S=QK^T), V^T [64][224] (= transposed-B for P@V),
// per-wave 16x224 fp32 score rows. ~194 KB dynamic LDS (WGP has 320 KB).

#define LDS_QS_OFF 0
#define LDS_KS_OFF (208 * 64 * 2)                       // 26624
#define LDS_VT_OFF (LDS_KS_OFF + NPAD * 64 * 2)         // 55296
#define LDS_PS_OFF (LDS_VT_OFF + 64 * NPAD * 2)         // 83968
#define LDS_TOTAL  (LDS_PS_OFF + 8 * 16 * NPAD * 4)     // 198656

__global__ void spatial_attn(const _Float16* __restrict__ Qh,
                             const _Float16* __restrict__ Kh,
                             const _Float16* __restrict__ Vt,   // [g][d][208]
                             float* __restrict__ attnOut,
                             _Float16* __restrict__ outS) {
    extern __shared__ __align__(16) char smem[];
    _Float16* Qs  = (_Float16*)(smem + LDS_QS_OFF);  // [208][64]
    _Float16* Ks  = (_Float16*)(smem + LDS_KS_OFF);  // [224][64] (x-major)
    _Float16* Vst = (_Float16*)(smem + LDS_VT_OFF);  // [64][224] (d-major)
    float*    Ps  = (float*)   (smem + LDS_PS_OFF);  // [8][16][224]

    const int g = blockIdx.x;           // (h*16 + b)*8 + t
    const int t = g & 7;
    const int b = (g >> 3) & 15;
    const int h = g >> 7;
    const int tid  = threadIdx.x;
    const int wave = tid >> 5;
    const int lane = tid & 31;
    const int half = lane >> 4;
    const int mrow = lane & 15;
    const int4 zero4 = make_int4(0, 0, 0, 0);

    // ---- cooperative stage-in, all 16-byte chunks ----
    // Qs[w][d]: 208 rows x 8 int4
    for (int idx = tid; idx < 208 * 8; idx += 256) {
        int w = idx >> 3, c = idx & 7;
        int4 v = zero4;
        if (w < WH1)
            v = *(const int4*)(Qh + ((size_t)((t * WH1 + w) * B_DIM + b)) * C_DIM
                               + h * DH + c * 8);
        ((int4*)Qs)[idx] = v;
    }
    // Ks[x][d]: 224 rows x 8 int4
    for (int idx = tid; idx < NPAD * 8; idx += 256) {
        int x = idx >> 3, c = idx & 7;
        int4 v = zero4;
        if (x < WH1)
            v = *(const int4*)(Kh + ((size_t)((t * WH1 + x) * B_DIM + b)) * C_DIM
                               + h * DH + c * 8);
        ((int4*)Ks)[idx] = v;
    }
    // Vst[d][x]: 64 rows x 28 int4 (chunks >=26 are past the 208-wide source)
    for (int idx = tid; idx < 64 * 28; idx += 256) {
        int d = idx / 28, c = idx % 28;
        int4 v = zero4;
        if (c < 26)  // covers x 0..207; x>=197 is junk but P zeros those cols
            v = *(const int4*)(Vt + (size_t)g * DH * XPAD + (size_t)d * XPAD
                               + c * 8);
        ((int4*)Vst)[idx] = v;
    }
    __syncthreads();

    float* P = Ps + wave * 16 * NPAD;   // this wave's 16x224 score rows

    for (int round = 0; round < 2; ++round) {
        int rb = wave + round * 8;      // 13 row blocks over 8 waves
        bool active = rb < 13;          // wave-uniform
        int row0 = rb * 16;

        if (active) {                   // S = Q K^T / sqrt(d)
            v16h a0 = load_frag16(Qs, 64, row0 + mrow, 0,  half);
            v16h a1 = load_frag16(Qs, 64, row0 + mrow, 32, half);
            for (int ct = 0; ct < 14; ++ct) {
                v8f acc = {};
                v16h b0 = load_frag16(Ks, 64, ct * 16 + mrow, 0,  half);
                acc = wmma_f16(a0, b0, acc);
                v16h b1 = load_frag16(Ks, 64, ct * 16 + mrow, 32, half);
                acc = wmma_f16(a1, b1, acc);
#pragma unroll
                for (int i = 0; i < 8; ++i)
                    P[(i + 8 * half) * NPAD + ct * 16 + mrow] = acc[i] * 0.125f;
            }
        }
        __syncthreads();

        if (active) {  // row softmax: 2 lanes per row, 112 cols each
            int r = mrow, grow = row0 + r;
            int x0 = half * 112, x1 = x0 + 112;
            float m = -1e30f;
            for (int x = x0; x < x1; ++x) {
                float s = (x < WH1) ? P[r * NPAD + x] : -1e30f;
                m = fmaxf(m, s);
            }
            m = fmaxf(m, __shfl_xor(m, 16, 32));
            float sum = 0.f;
            for (int x = x0; x < x1; ++x) {
                float e = 0.f;
                if (x < WH1) e = __expf(P[r * NPAD + x] - m);
                P[r * NPAD + x] = e;   // pad cols become 0 here
                sum += e;
            }
            sum += __shfl_xor(sum, 16, 32);
            float inv = 1.f / sum;
            for (int x = x0; x < x1; ++x) {
                float p = P[r * NPAD + x] * inv;
                P[r * NPAD + x] = p;
                if (grow < WH1 && x < WH1)
                    attnOut[(size_t)g * WH1 * WH1 + (size_t)grow * WH1 + x] = p;
            }
        }
        __syncthreads();

        if (active) {                   // O = P @ V  (16x224 @ 224x64)
            v8f o0 = {}, o1 = {}, o2 = {}, o3 = {};
            for (int ks = 0; ks < NPAD; ks += 32) {
                v16h a = load_frag16_f32(P, NPAD, mrow, ks, half);
                v16h bb;
                bb = load_frag16(Vst, NPAD,  0 + mrow, ks, half); o0 = wmma_f16(a, bb, o0);
                bb = load_frag16(Vst, NPAD, 16 + mrow, ks, half); o1 = wmma_f16(a, bb, o1);
                bb = load_frag16(Vst, NPAD, 32 + mrow, ks, half); o2 = wmma_f16(a, bb, o2);
                bb = load_frag16(Vst, NPAD, 48 + mrow, ks, half); o3 = wmma_f16(a, bb, o3);
            }
#pragma unroll
            for (int ct = 0; ct < 4; ++ct) {
                v8f acc = (ct == 0) ? o0 : (ct == 1) ? o1 : (ct == 2) ? o2 : o3;
#pragma unroll
                for (int i = 0; i < 8; ++i) {
                    int grow = row0 + i + 8 * half;
                    int d = ct * 16 + mrow;
                    if (grow < WH1)
                        outS[((size_t)((t * WH1 + grow) * B_DIM + b)) * C_DIM
                             + h * DH + d] = (_Float16)acc[i];
                }
            }
        }
        __syncthreads();
    }
}

// -------------------- host orchestration ----------------------------------

extern "C" void kernel_launch(void* const* d_in, const int* in_sizes, int n_in,
                              void* d_out, int out_size, void* d_ws,
                              size_t ws_size, hipStream_t stream) {
    (void)in_sizes; (void)n_in; (void)out_size; (void)ws_size;

    // inputs: 0=q 1=k(unused) 2=value(unused) 3=orig_shape(unused)
    //         4=in_proj_weight 5=in_proj_bias 6=out_w 7=out_b
    const float* q     = (const float*)d_in[0];
    const float* w_in  = (const float*)d_in[4];
    const float* b_in  = (const float*)d_in[5];
    const float* w_out = (const float*)d_in[6];
    const float* b_out = (const float*)d_in[7];

    float* outMain = (float*)d_out;                         // 25216 x 768
    float* attnOut = outMain + (size_t)MROWS * C_DIM;       // 1536 x 197 x 197

    char* ws = (char*)d_ws;
    size_t off = 0;
    auto alloc = [&](size_t bytes) -> void* {
        void* p = ws + off;
        off += (bytes + 255) & ~(size_t)255;
        return p;
    };
    _Float16* qh    = (_Float16*)alloc((size_t)MROWS * C_DIM * 2);
    _Float16* winH  = (_Float16*)alloc((size_t)3 * C_DIM * C_DIM * 2); // [2304][768]
    _Float16* woutH = (_Float16*)alloc((size_t)C_DIM * C_DIM * 2);     // [768][768]
    _Float16* Qh    = (_Float16*)alloc((size_t)MROWS * C_DIM * 2);
    _Float16* Kh    = (_Float16*)alloc((size_t)MROWS * C_DIM * 2);
    _Float16* Vh    = (_Float16*)alloc((size_t)MROWS * C_DIM * 2);
    _Float16* outT  = (_Float16*)alloc((size_t)G_SPA * DH * XPAD * 2); // [g][d][208]
    _Float16* outS  = (_Float16*)alloc((size_t)MROWS * C_DIM * 2);

    int nq = MROWS * C_DIM;
    cvt_f32_to_f16<<<(nq + 255) / 256, 256, 0, stream>>>(q, qh, nq);
    cvt_f32_to_f16<<<(3 * C_DIM * C_DIM + 255) / 256, 256, 0, stream>>>(
        w_in, winH, 3 * C_DIM * C_DIM);
    cvt_f32_to_f16<<<(C_DIM * C_DIM + 255) / 256, 256, 0, stream>>>(
        w_out, woutH, C_DIM * C_DIM);

    // QKV projection: 25216 x 2304 (B = winH^T implicitly via [N][K] layout)
    gemm_wmma_f16<<<dim3(3 * C_DIM / 64, MROWS / 128), 256, 0, stream>>>(
        qh, winH, b_in, nullptr, Qh, Kh, Vh, MROWS, 3 * C_DIM, C_DIM, 1);

    temporal_attn<<<NHEAD * B_DIM * WH1, 64, 0, stream>>>(Qh, Kh, Vh, outT);

    spatial_attn<<<G_SPA, 256, LDS_TOTAL, stream>>>(Qh, Kh, outT, attnOut, outS);

    // output projection: 25216 x 768
    gemm_wmma_f16<<<dim3(C_DIM / 64, MROWS / 128), 256, 0, stream>>>(
        outS, woutH, b_out, outMain, nullptr, nullptr, nullptr,
        MROWS, C_DIM, C_DIM, 0);
}
